// PoseVRNN_16595753632537
// MI455X (gfx1250) — matrix operand order repaired and checked by reading, not compiled, not run on hardware
//
#include <hip/hip_runtime.h>
#include <cstddef>

typedef __attribute__((ext_vector_type(16))) _Float16 v16h;
typedef __attribute__((ext_vector_type(8)))  _Float16 v8h;
typedef __attribute__((ext_vector_type(8)))  float    v8f;

#define WMMA_F16(acc, A, B) \
  (acc) = __builtin_amdgcn_wmma_f32_16x16x32_f16(false, (A), false, (B), (short)0, (acc), false, false)

// ---------------- geometry ----------------
constexpr int POSE = 4, H = 64, Z = 64, B_SZ = 2048, T = 128;
constexpr int WAVES = 4;                               // waves per workgroup
constexpr int ROWS_PER_WAVE = 16;
constexpr int NBLK = B_SZ / (ROWS_PER_WAVE * WAVES);   // 32 workgroups
constexpr int NWAVES_TOTAL = B_SZ / ROWS_PER_WAVE;     // 128 waves

// ---------------- weight fragment table (1 frag = 512 halfs = 1 KB) ----------------
constexpr int F_PX = 0,  F_PZ = 4,  F_ENC = 12, F_EM = 28, F_ES = 36,
              F_PR = 44, F_PRM = 52, F_PRS = 60, F_DEC = 68, F_DM = 84,
              F_IH0 = 86, F_HH0 = 134, F_IH1 = 158, F_HH1 = 182, N_FRAGS = 206;
constexpr int FRAG_HALFS = 512;
constexpr int W_HALFS = N_FRAGS * FRAG_HALFS;          // 105472 halfs
constexpr int W_BYTES = W_HALFS * 2;                   // 210944 B

// ---------------- bias table (floats, zero-padded) ----------------
constexpr int B_PX = 0, B_PZ = 64, B_ENC = 128, B_EM = 192, B_ES = 256, B_PR = 320,
              B_PRM = 384, B_PRS = 448, B_DEC = 512, B_DM = 576, B_IH0 = 592,
              B_HH0 = 784, B_IH1 = 976, B_HH1 = 1168, N_BIAS = 1360;
constexpr int BIAS_BYTES = N_BIAS * 4;                 // 5440 B
constexpr int CONST_BYTES = W_BYTES + BIAS_BYTES;      // 216384 B

// ---------------- per-wave LDS scratch (offsets in halfs) ----------------
constexpr int S_XPAD = 0, S_PHIX = 512, S_PHIZ = 1536, S_ENC = 2560, S_PRI = 3584,
              S_ZDEC = 4608, S_H0A = 5632, S_H0B = 6656, S_H1A = 7680, S_H1B = 8704,
              SCR_HALFS = 9728;
constexpr int SMEM_BYTES = CONST_BYTES + WAVES * SCR_HALFS * 2;   // 294208 B
static_assert(SMEM_BYTES <= 320 * 1024, "LDS budget");

// ------------------------------------------------------------------
// Kernel 0: swizzle one weight matrix (row-major f32 KxN) into B-fragment
// layout f16.  One block per fragment; thread t -> lane = t>>4, j = t&15.
// B fragment element j of lane L holds W[kt*32 + (j<8 ? (L/16)*8 + j : 16 + (L/16)*8 + j-8)][nt*16 + (L%16)].
// ------------------------------------------------------------------
__global__ void swizzle_w(const float* __restrict__ W, _Float16* __restrict__ dst,
                          int K, int N, int ntiles) {
  int frag = blockIdx.x;
  int kt = frag / ntiles, nt = frag % ntiles;
  int t = threadIdx.x;
  int lane = t >> 4, j = t & 15;
  int k = kt * 32 + (j < 8 ? (lane >> 4) * 8 + j : 16 + (lane >> 4) * 8 + (j - 8));
  int n = nt * 16 + (lane & 15);
  float v = (k < K && n < N) ? W[(size_t)k * N + n] : 0.0f;
  dst[(size_t)frag * FRAG_HALFS + lane * 16 + j] = (_Float16)v;
}

// ------------------------------------------------------------------
// Kernel 1: pack all bias vectors into one padded f32 table.
// ------------------------------------------------------------------
__global__ void pack_bias(float* __restrict__ dst,
    const float* bpx, const float* bpz, const float* benc, const float* bem,
    const float* bes, const float* bpr, const float* bprm, const float* bprs,
    const float* bdec, const float* bdm, const float* bih0, const float* bhh0,
    const float* bih1, const float* bhh1) {
  int i = blockIdx.x * blockDim.x + threadIdx.x;
  if (i >= N_BIAS) return;
  float v = 0.f;
  if      (i < B_PZ)          v = bpx[i - B_PX];
  else if (i < B_ENC)         v = bpz[i - B_PZ];
  else if (i < B_EM)          v = benc[i - B_ENC];
  else if (i < B_ES)          v = bem[i - B_EM];
  else if (i < B_PR)          v = bes[i - B_ES];
  else if (i < B_PRM)         v = bpr[i - B_PR];
  else if (i < B_PRS)         v = bprm[i - B_PRM];
  else if (i < B_DEC)         v = bprs[i - B_PRS];
  else if (i < B_DM)          v = bdec[i - B_DEC];
  else if (i < B_DM + POSE)   v = bdm[i - B_DM];
  else if (i < B_IH0)         v = 0.f;                 // pad
  else if (i < B_HH0)         v = bih0[i - B_IH0];
  else if (i < B_IH1)         v = bhh0[i - B_HH0];
  else if (i < B_HH1)         v = bih1[i - B_IH1];
  else                        v = bhh1[i - B_HH1];
  dst[i] = v;
}

// ------------------------------------------------------------------
// device helpers (wave32 WMMA fragment plumbing through per-wave LDS)
// ------------------------------------------------------------------
__device__ __forceinline__ v16h ldA(const _Float16* buf, int stride, int kb) {
  int lane = threadIdx.x & 31;
  const _Float16* p = buf + (lane & 15) * stride + kb + ((lane >> 4) << 3);
  union { v16h v; v8h h[2]; } u;
  u.h[0] = *(const v8h*)p;          // k = kb+hioff .. +7   (ds_load_b128)
  u.h[1] = *(const v8h*)(p + 16);   // k = kb+16+hioff .. +7
  return u.v;
}
__device__ __forceinline__ v16h ldB(const _Float16* sw, int frag) {
  int lane = threadIdx.x & 31;
  return *(const v16h*)(sw + (size_t)frag * FRAG_HALFS + lane * 16);
}
__device__ __forceinline__ v8f bsplat(const float* sbias, int off) {
  float b = sbias[off + (threadIdx.x & 15)];
  v8f a;
#pragma unroll
  for (int r = 0; r < 8; ++r) a[r] = b;
  return a;
}
__device__ __forceinline__ void stC(v8f c, _Float16* buf, int stride, int nb, bool relu) {
  int lane = threadIdx.x & 31;
  int n = lane & 15, mo = (lane >> 4) * 8;
#pragma unroll
  for (int r = 0; r < 8; ++r) {
    float v = c[r];
    if (relu) v = fmaxf(v, 0.f);            // single v_max_num_f32
    buf[(mo + r) * stride + nb + n] = (_Float16)v;
  }
}
__device__ __forceinline__ v8f ldC(const _Float16* buf, int stride, int nb) {
  int lane = threadIdx.x & 31;
  int n = lane & 15, mo = (lane >> 4) * 8;
  v8f a;
#pragma unroll
  for (int r = 0; r < 8; ++r) a[r] = (float)buf[(mo + r) * stride + nb + n];
  return a;
}
// Fast transcendentals built from v_exp_f32 / v_log_f32 / v_rcp_f32 so they
// stay inline VALU/TRANS (co-executes with XDL WMMA) instead of ocml libcalls.
__device__ __forceinline__ float softplusf(float v) {
  return v > 15.f ? v : __logf(1.f + __expf(v));
}
__device__ __forceinline__ float sigmoidf(float v) {
  return __builtin_amdgcn_rcpf(1.f + __expf(-v));
}
__device__ __forceinline__ float tanh_fast(float v) {
  float e2 = __expf(2.f * v);
  return (e2 - 1.f) * __builtin_amdgcn_rcpf(e2 + 1.f);
}

// ------------------------------------------------------------------
// Kernel 2: main VRNN. One wave = 16 batch rows, full 127-step recurrence.
// ------------------------------------------------------------------
__global__ __launch_bounds__(WAVES * 32) void vrnn_main(
    const float* __restrict__ x, const float* __restrict__ eps,
    const uint4* __restrict__ wsConst, float* __restrict__ partials,
    float* __restrict__ out) {
  extern __shared__ __align__(16) char smem[];
  _Float16* sw = (_Float16*)smem;
  const float* sbias = (const float*)(smem + W_BYTES);

  // cooperative copy: weights (f16 fragments) + biases -> LDS, once
  {
    uint4* d = (uint4*)smem;
    for (int i = threadIdx.x; i < CONST_BYTES / 16; i += blockDim.x) d[i] = wsConst[i];
  }
  __syncthreads();

  const int lane = threadIdx.x & 31;
  const int wid  = threadIdx.x >> 5;
  const int gwid = blockIdx.x * WAVES + wid;
  const int row0 = gwid * ROWS_PER_WAVE;

  _Float16* scr  = (_Float16*)(smem + CONST_BYTES) + wid * SCR_HALFS;
  _Float16* xpad = scr + S_XPAD;
  _Float16* phix = scr + S_PHIX;
  _Float16* phiz = scr + S_PHIZ;
  _Float16* encb = scr + S_ENC;
  _Float16* prib = scr + S_PRI;
  _Float16* zdec = scr + S_ZDEC;
  _Float16* h0c = scr + S_H0A; _Float16* h0n = scr + S_H0B;
  _Float16* h1c = scr + S_H1A; _Float16* h1n = scr + S_H1B;

  // zero per-wave scratch (xpad pad columns + initial hidden state)
  for (int i = lane * 8; i < SCR_HALFS; i += 32 * 8) { v8h zv = {}; *(v8h*)(scr + i) = zv; }

  const int n  = lane & 15;
  const int mo = (lane >> 4) * 8;

  float kld = 0.f, mse = 0.f;

  for (int t = 0; t < T - 1; ++t) {
    // ---- x_t tile -> xpad (16x32, cols 4..31 stay zero) ----
    if (lane < 16) {
      const float4 xv = *(const float4*)(x + ((size_t)(row0 + lane) * T + t) * POSE);
      _Float16* d = xpad + lane * 32;
      d[0] = (_Float16)xv.x; d[1] = (_Float16)xv.y;
      d[2] = (_Float16)xv.z; d[3] = (_Float16)xv.w;
    }
    // ---- phi_x = relu(x @ Wpx + bpx) ----
    {
      v16h a = ldA(xpad, 32, 0);
#pragma unroll
      for (int nt = 0; nt < 4; ++nt) {
        v8f acc = bsplat(sbias, B_PX + nt * 16);
        WMMA_F16(acc, a, ldB(sw, F_PX + nt));
        stC(acc, phix, 64, nt * 16, true);
      }
    }
    // ---- h_top A fragments (reused by enc, pri, dec, GRU1) ----
    v16h ht0 = ldA(h1c, 64, 0), ht1 = ldA(h1c, 64, 32);
    // ---- enc = relu([phi_x, h_top] @ Wenc + benc) ----
    {
      v16h a0 = ldA(phix, 64, 0), a1 = ldA(phix, 64, 32);
#pragma unroll
      for (int nt = 0; nt < 4; ++nt) {
        v8f acc = bsplat(sbias, B_ENC + nt * 16);
        WMMA_F16(acc, a0,  ldB(sw, F_ENC + 0  + nt));
        WMMA_F16(acc, a1,  ldB(sw, F_ENC + 4  + nt));
        WMMA_F16(acc, ht0, ldB(sw, F_ENC + 8  + nt));
        WMMA_F16(acc, ht1, ldB(sw, F_ENC + 12 + nt));
        stC(acc, encb, 64, nt * 16, true);
      }
    }
    // ---- pri = relu(h_top @ Wpr + bpr) ----
#pragma unroll
    for (int nt = 0; nt < 4; ++nt) {
      v8f acc = bsplat(sbias, B_PR + nt * 16);
      WMMA_F16(acc, ht0, ldB(sw, F_PR + nt));
      WMMA_F16(acc, ht1, ldB(sw, F_PR + 4 + nt));
      stC(acc, prib, 64, nt * 16, true);
    }
    // ---- enc_mean/std, pri_mean/std, z, KLD ----
    {
      v16h e0 = ldA(encb, 64, 0), e1 = ldA(encb, 64, 32);
      v16h p0 = ldA(prib, 64, 0), p1 = ldA(prib, 64, 32);
#pragma unroll
      for (int nt = 0; nt < 4; ++nt) {
        v8f em = bsplat(sbias, B_EM + nt * 16);
        WMMA_F16(em, e0, ldB(sw, F_EM + nt));  WMMA_F16(em, e1, ldB(sw, F_EM + 4 + nt));
        v8f es = bsplat(sbias, B_ES + nt * 16);
        WMMA_F16(es, e0, ldB(sw, F_ES + nt));  WMMA_F16(es, e1, ldB(sw, F_ES + 4 + nt));
        v8f pm = bsplat(sbias, B_PRM + nt * 16);
        WMMA_F16(pm, p0, ldB(sw, F_PRM + nt)); WMMA_F16(pm, p1, ldB(sw, F_PRM + 4 + nt));
        v8f ps = bsplat(sbias, B_PRS + nt * 16);
        WMMA_F16(ps, p0, ldB(sw, F_PRS + nt)); WMMA_F16(ps, p1, ldB(sw, F_PRS + 4 + nt));
#pragma unroll
        for (int r = 0; r < 8; ++r) {
          float se = softplusf(es[r]);
          float sp = softplusf(ps[r]);
          float ev = eps[((size_t)t * B_SZ + row0 + mo + r) * Z + nt * 16 + n];
          float zv = em[r] + ev * se;
          zdec[(mo + r) * 64 + nt * 16 + n] = (_Float16)zv;
          float dm = em[r] - pm[r];
          kld += 0.5f * (2.f * (__logf(sp) - __logf(se)) +
                         (se * se + dm * dm) * __builtin_amdgcn_rcpf(sp * sp) - 1.f);
        }
      }
    }
    // ---- phi_z = relu(z @ Wpz + bpz) ----
    {
      v16h z0 = ldA(zdec, 64, 0), z1 = ldA(zdec, 64, 32);
#pragma unroll
      for (int nt = 0; nt < 4; ++nt) {
        v8f acc = bsplat(sbias, B_PZ + nt * 16);
        WMMA_F16(acc, z0, ldB(sw, F_PZ + nt));
        WMMA_F16(acc, z1, ldB(sw, F_PZ + 4 + nt));
        stC(acc, phiz, 64, nt * 16, true);
      }
    }
    // ---- dec = relu([phi_z, h_top] @ Wdec + bdec) (overwrites z buffer) ----
    {
      v16h a0 = ldA(phiz, 64, 0), a1 = ldA(phiz, 64, 32);
#pragma unroll
      for (int nt = 0; nt < 4; ++nt) {
        v8f acc = bsplat(sbias, B_DEC + nt * 16);
        WMMA_F16(acc, a0,  ldB(sw, F_DEC + nt));
        WMMA_F16(acc, a1,  ldB(sw, F_DEC + 4 + nt));
        WMMA_F16(acc, ht0, ldB(sw, F_DEC + 8 + nt));
        WMMA_F16(acc, ht1, ldB(sw, F_DEC + 12 + nt));
        stC(acc, zdec, 64, nt * 16, true);
      }
    }
    // ---- pred = dec @ Wdm + bdm ; MSE ; store preds ----
    {
      v16h d0 = ldA(zdec, 64, 0), d1 = ldA(zdec, 64, 32);
      v8f acc = bsplat(sbias, B_DM);
      WMMA_F16(acc, d0, ldB(sw, F_DM + 0));
      WMMA_F16(acc, d1, ldB(sw, F_DM + 1));
      if (n < POSE) {
#pragma unroll
        for (int r = 0; r < 8; ++r) {
          int m = row0 + mo + r;
          float xn = x[((size_t)m * T + t + 1) * POSE + n];
          float d  = xn - acc[r];
          mse += d * d;
          out[2 + ((size_t)m * (T - 1) + t) * POSE + n] = acc[r];
        }
      }
    }
    // ---- GRU layer 0: in = [phi_x, phi_z] (K=128), hidden h0 (K=64) ----
    {
      v16h px0 = ldA(phix, 64, 0), px1 = ldA(phix, 64, 32);
      v16h pz0 = ldA(phiz, 64, 0), pz1 = ldA(phiz, 64, 32);
      v16h h00 = ldA(h0c, 64, 0),  h01 = ldA(h0c, 64, 32);
#pragma unroll
      for (int g = 0; g < 4; ++g) {
        v8f ir = bsplat(sbias, B_IH0 + g * 16);
        WMMA_F16(ir, px0, ldB(sw, F_IH0 + 0  + g));
        WMMA_F16(ir, px1, ldB(sw, F_IH0 + 12 + g));
        WMMA_F16(ir, pz0, ldB(sw, F_IH0 + 24 + g));
        WMMA_F16(ir, pz1, ldB(sw, F_IH0 + 36 + g));
        v8f iz = bsplat(sbias, B_IH0 + 64 + g * 16);
        WMMA_F16(iz, px0, ldB(sw, F_IH0 + 0  + g + 4));
        WMMA_F16(iz, px1, ldB(sw, F_IH0 + 12 + g + 4));
        WMMA_F16(iz, pz0, ldB(sw, F_IH0 + 24 + g + 4));
        WMMA_F16(iz, pz1, ldB(sw, F_IH0 + 36 + g + 4));
        v8f in_ = bsplat(sbias, B_IH0 + 128 + g * 16);
        WMMA_F16(in_, px0, ldB(sw, F_IH0 + 0  + g + 8));
        WMMA_F16(in_, px1, ldB(sw, F_IH0 + 12 + g + 8));
        WMMA_F16(in_, pz0, ldB(sw, F_IH0 + 24 + g + 8));
        WMMA_F16(in_, pz1, ldB(sw, F_IH0 + 36 + g + 8));
        v8f hr = bsplat(sbias, B_HH0 + g * 16);
        WMMA_F16(hr, h00, ldB(sw, F_HH0 + g));
        WMMA_F16(hr, h01, ldB(sw, F_HH0 + 12 + g));
        v8f hz = bsplat(sbias, B_HH0 + 64 + g * 16);
        WMMA_F16(hz, h00, ldB(sw, F_HH0 + g + 4));
        WMMA_F16(hz, h01, ldB(sw, F_HH0 + 12 + g + 4));
        v8f hn = bsplat(sbias, B_HH0 + 128 + g * 16);
        WMMA_F16(hn, h00, ldB(sw, F_HH0 + g + 8));
        WMMA_F16(hn, h01, ldB(sw, F_HH0 + 12 + g + 8));
        v8f hold = ldC(h0c, 64, g * 16);
#pragma unroll
        for (int r = 0; r < 8; ++r) {
          float rg = sigmoidf(ir[r] + hr[r]);
          float zg = sigmoidf(iz[r] + hz[r]);
          float ng = tanh_fast(in_[r] + rg * hn[r]);
          float hv = (1.f - zg) * ng + zg * hold[r];
          h0n[(mo + r) * 64 + g * 16 + n] = (_Float16)hv;
        }
      }
    }
    // ---- GRU layer 1: in = h0_new (K=64), hidden h1 (K=64) ----
    {
      v16h a0 = ldA(h0n, 64, 0), a1 = ldA(h0n, 64, 32);
#pragma unroll
      for (int g = 0; g < 4; ++g) {
        v8f ir = bsplat(sbias, B_IH1 + g * 16);
        WMMA_F16(ir, a0, ldB(sw, F_IH1 + g));
        WMMA_F16(ir, a1, ldB(sw, F_IH1 + 12 + g));
        v8f iz = bsplat(sbias, B_IH1 + 64 + g * 16);
        WMMA_F16(iz, a0, ldB(sw, F_IH1 + g + 4));
        WMMA_F16(iz, a1, ldB(sw, F_IH1 + 12 + g + 4));
        v8f in_ = bsplat(sbias, B_IH1 + 128 + g * 16);
        WMMA_F16(in_, a0, ldB(sw, F_IH1 + g + 8));
        WMMA_F16(in_, a1, ldB(sw, F_IH1 + 12 + g + 8));
        v8f hr = bsplat(sbias, B_HH1 + g * 16);
        WMMA_F16(hr, ht0, ldB(sw, F_HH1 + g));
        WMMA_F16(hr, ht1, ldB(sw, F_HH1 + 12 + g));
        v8f hz = bsplat(sbias, B_HH1 + 64 + g * 16);
        WMMA_F16(hz, ht0, ldB(sw, F_HH1 + g + 4));
        WMMA_F16(hz, ht1, ldB(sw, F_HH1 + 12 + g + 4));
        v8f hn = bsplat(sbias, B_HH1 + 128 + g * 16);
        WMMA_F16(hn, ht0, ldB(sw, F_HH1 + g + 8));
        WMMA_F16(hn, ht1, ldB(sw, F_HH1 + 12 + g + 8));
        v8f hold = ldC(h1c, 64, g * 16);
#pragma unroll
        for (int r = 0; r < 8; ++r) {
          float rg = sigmoidf(ir[r] + hr[r]);
          float zg = sigmoidf(iz[r] + hz[r]);
          float ng = tanh_fast(in_[r] + rg * hn[r]);
          float hv = (1.f - zg) * ng + zg * hold[r];
          h1n[(mo + r) * 64 + g * 16 + n] = (_Float16)hv;
        }
      }
    }
    // ---- double-buffer swap ----
    { _Float16* tp = h0c; h0c = h0n; h0n = tp;
      tp = h1c; h1c = h1n; h1n = tp; }
  }

  // wave reduction of partial kld / mse (wave32)
#pragma unroll
  for (int o = 16; o > 0; o >>= 1) {
    kld += __shfl_xor(kld, o, 32);
    mse += __shfl_xor(mse, o, 32);
  }
  if (lane == 0) {
    partials[gwid * 2 + 0] = kld;
    partials[gwid * 2 + 1] = mse;
  }
}

// ------------------------------------------------------------------
// Kernel 3: deterministic fixed-order final reduction.
// ------------------------------------------------------------------
__global__ void finalize(const float* __restrict__ partials, float* __restrict__ out) {
  if (threadIdx.x == 0 && blockIdx.x == 0) {
    float k = 0.f, m = 0.f;
    for (int i = 0; i < NWAVES_TOTAL; ++i) { k += partials[2 * i]; m += partials[2 * i + 1]; }
    out[0] = k;
    out[1] = m * (1.f / (float)(B_SZ * POSE));
  }
}

// ------------------------------------------------------------------
extern "C" void kernel_launch(void* const* d_in, const int* in_sizes, int n_in,
                              void* d_out, int out_size, void* d_ws, size_t ws_size,
                              hipStream_t stream) {
  (void)in_sizes; (void)n_in; (void)out_size; (void)ws_size;
  const float* x    = (const float*)d_in[0];
  const float* eps  = (const float*)d_in[1];
  const float* Wpx  = (const float*)d_in[2];  const float* bpx  = (const float*)d_in[3];
  const float* Wpz  = (const float*)d_in[4];  const float* bpz  = (const float*)d_in[5];
  const float* Wenc = (const float*)d_in[6];  const float* benc = (const float*)d_in[7];
  const float* Wem  = (const float*)d_in[8];  const float* bem  = (const float*)d_in[9];
  const float* Wes  = (const float*)d_in[10]; const float* bes  = (const float*)d_in[11];
  const float* Wpr  = (const float*)d_in[12]; const float* bpr  = (const float*)d_in[13];
  const float* Wprm = (const float*)d_in[14]; const float* bprm = (const float*)d_in[15];
  const float* Wprs = (const float*)d_in[16]; const float* bprs = (const float*)d_in[17];
  const float* Wdec = (const float*)d_in[18]; const float* bdec = (const float*)d_in[19];
  const float* Wdm  = (const float*)d_in[20]; const float* bdm  = (const float*)d_in[21];
  const float* Wih0 = (const float*)d_in[22]; const float* Whh0 = (const float*)d_in[23];
  const float* bih0 = (const float*)d_in[24]; const float* bhh0 = (const float*)d_in[25];
  const float* Wih1 = (const float*)d_in[26]; const float* Whh1 = (const float*)d_in[27];
  const float* bih1 = (const float*)d_in[28]; const float* bhh1 = (const float*)d_in[29];

  _Float16* wsW = (_Float16*)d_ws;
  float* wsB = (float*)((char*)d_ws + W_BYTES);
  float* wsP = (float*)((char*)d_ws + CONST_BYTES);
  float* out = (float*)d_out;

  auto swz = [&](const float* W, int fbase, int K, int N) {
    int kt = (K + 31) / 32, nt = (N + 15) / 16;
    swizzle_w<<<kt * nt, 512, 0, stream>>>(W, wsW + (size_t)fbase * FRAG_HALFS, K, N, nt);
  };
  swz(Wpx,  F_PX,  POSE,  H);
  swz(Wpz,  F_PZ,  Z,     H);
  swz(Wenc, F_ENC, 2 * H, H);
  swz(Wem,  F_EM,  H,     Z);
  swz(Wes,  F_ES,  H,     Z);
  swz(Wpr,  F_PR,  H,     H);
  swz(Wprm, F_PRM, H,     Z);
  swz(Wprs, F_PRS, H,     Z);
  swz(Wdec, F_DEC, 2 * H, H);
  swz(Wdm,  F_DM,  H,     POSE);
  swz(Wih0, F_IH0, 2 * H, 3 * H);
  swz(Whh0, F_HH0, H,     3 * H);
  swz(Wih1, F_IH1, H,     3 * H);
  swz(Whh1, F_HH1, H,     3 * H);

  pack_bias<<<(N_BIAS + 255) / 256, 256, 0, stream>>>(
      wsB, bpx, bpz, benc, bem, bes, bpr, bprm, bprs, bdec, bdm,
      bih0, bhh0, bih1, bhh1);

  (void)hipFuncSetAttribute(reinterpret_cast<const void*>(vrnn_main),
                            hipFuncAttributeMaxDynamicSharedMemorySize, SMEM_BYTES);
  vrnn_main<<<NBLK, WAVES * 32, SMEM_BYTES, stream>>>(
      x, eps, (const uint4*)d_ws, wsP, out);

  finalize<<<1, 32, 0, stream>>>(wsP, out);
}
